// InvertedResidualBlock_3324304687836
// MI455X (gfx1250) — matrix operand order repaired and checked by reading, not compiled, not run on hardware
//
#include <hip/hip_runtime.h>

// ---------------- CDNA5 (gfx1250) fused inverted-residual block ----------------
// wave32, WMMA v_wmma_f32_16x16x32_f16, full block fused through 320KB LDS.

typedef __attribute__((ext_vector_type(16))) _Float16 v16h;
typedef __attribute__((ext_vector_type(8)))  float    v8f;

#define CIN  96
#define HDM  576
#define IMG  56
#define TH   8          // output tile height
#define TW   8          // output tile width
#define HX   10         // halo width  (TW+2)
#define HP   100        // halo pixels (10*10)
#define NT1  7          // ceil(100/16) pixel tiles, stage 1
#define MT1  36         // 576/16 channel tiles, stage 1
#define KC1  3          // 96/32  K chunks, stage 1
#define NTHREADS 256    // 8 waves

union HV  { v16h v; unsigned int u[8]; _Float16 h[16]; };
union H2  { unsigned int u; _Float16 h[2]; };

// Gather a 16x32 f16 A-tile from an fp32 weight row, per the CDNA5 16-bit
// A-matrix VGPR layout: lane M = lane%16; half e -> K = e + (e>=8?8:0) + (lane>=16?8:0).
// K pairs are consecutive -> compiler merges to global_load_b128 + v_cvt_pk_f16_f32.
__device__ __forceinline__ v16h load_a_f32(const float* __restrict__ row, int lhi) {
  HV a;
#pragma unroll
  for (int e = 0; e < 16; e += 2) {
    const int K = e + ((e & 8) ? 8 : 0) + (lhi ? 8 : 0);
    const float2 f = *(const float2*)(row + K);
    a.h[e]     = (_Float16)f.x;
    a.h[e + 1] = (_Float16)f.y;
  }
  return a.v;
}

// A-tile from a row-major f16 LDS image: the lane's 16 halves are two
// contiguous 8-half K-runs (base+lhi*8 and base+16+lhi*8) -> 2x ds_load_b128.
__device__ __forceinline__ v16h load_a_lds(const _Float16* rowbase) {
  HV a;
  const unsigned int* p0 = (const unsigned int*)rowbase;
  const unsigned int* p1 = (const unsigned int*)(rowbase + 16);
#pragma unroll
  for (int q = 0; q < 4; ++q) { a.u[q] = p0[q]; a.u[4 + q] = p1[q]; }
  return a.v;
}

// Gather a 32x16 f16 B-tile from LDS stored [pixel][channel]:
// lane N = lane%16 (pixel), halves hold K = e + (e>=8?8:0) + 8*(lane/16) -> two
// 8-half runs (base, base+16), 16B aligned -> ds_load_b128 x2.
__device__ __forceinline__ v16h load_b_lds(const _Float16* base) {
  HV b;
  const unsigned int* p0 = (const unsigned int*)base;
  const unsigned int* p1 = (const unsigned int*)(base + 16);
#pragma unroll
  for (int q = 0; q < 4; ++q) { b.u[q] = p0[q]; b.u[4 + q] = p1[q]; }
  return b.v;
}

__global__ __launch_bounds__(NTHREADS)
void fused_ir_block(const float* __restrict__ xg,
                    const float* __restrict__ w1,
                    const float* __restrict__ g1, const float* __restrict__ b1,
                    const float* __restrict__ m1, const float* __restrict__ v1,
                    const float* __restrict__ wdw,
                    const float* __restrict__ gdw, const float* __restrict__ bdw,
                    const float* __restrict__ mdw, const float* __restrict__ vdw,
                    const float* __restrict__ w2,
                    const float* __restrict__ g2, const float* __restrict__ b2,
                    const float* __restrict__ m2, const float* __restrict__ v2,
                    float* __restrict__ outg)
{
  // LDS: 19200 + 115200 + 10368 + 4608 + 4608 + 768 = 154,752 B (2 WGs / 320KB WGP)
  __shared__ _Float16 xs[HP * CIN];     // x halo tile [pixel][ch]; later w2-f16 stage
  __shared__ _Float16 hs[HP * HDM];     // h / h2 tile,  [pixel][ch], f16
  __shared__ _Float16 wdws[9 * HDM];    // depthwise weights, [tap][ch], f16
  __shared__ float2   sb1[HDM];         // folded BN1 {scale,bias}
  __shared__ float2   sbd[HDM];         // folded BN-dw
  __shared__ float2   sb2[CIN];         // folded BN2

  const int tid  = threadIdx.x;
  const int lane = tid & 31;
  const int wv   = tid >> 5;      // wave id 0..7
  const int lmod = lane & 15;     // M (A rows) / N (B,D cols)
  const int lhi  = lane >> 4;     // upper half-wave flag

  const int n  = blockIdx.y;
  const int tY = blockIdx.x / (IMG / TW);
  const int tX = blockIdx.x % (IMG / TW);
  const int y0 = tY * TH - 1;     // halo origin in image coords
  const int x0 = tX * TW - 1;

  // ---------------- Stage 0: fold BN params, stage weights + x tile ----------
  for (int i = tid; i < HDM; i += NTHREADS) {
    float s = g1[i] * rsqrtf(v1[i] + 1e-5f);
    sb1[i] = make_float2(s, b1[i] - m1[i] * s);
    float sd = gdw[i] * rsqrtf(vdw[i] + 1e-5f);
    sbd[i] = make_float2(sd, bdw[i] - mdw[i] * sd);
  }
  for (int i = tid; i < CIN; i += NTHREADS) {
    float s = g2[i] * rsqrtf(v2[i] + 1e-5f);
    sb2[i] = make_float2(s, b2[i] - m2[i] * s);
  }
  for (int i = tid; i < 9 * HDM; i += NTHREADS) {   // transpose to [tap][ch]
    const int k = i / HDM, c = i % HDM;
    wdws[k * HDM + c] = (_Float16)wdw[c * 9 + k];
  }
  // x halo tile: zero-padded at image borders, stored [pixel][ch] f16
  for (int i = tid; i < HP * CIN; i += NTHREADS) {
    const int c = i / HP, p = i % HP;
    const int gy = y0 + p / HX, gx = x0 + p % HX;
    float val = 0.0f;
    if (gy >= 0 && gy < IMG && gx >= 0 && gx < IMG)
      val = xg[((size_t)(n * CIN + c) * IMG + gy) * IMG + gx];
    xs[p * CIN + c] = (_Float16)val;
  }
  __syncthreads();

  // ---------------- Stage 1: expand 1x1 GEMM + BN1 + ReLU -> hs --------------
  // h[576 x 100] = w1[576 x 96] * x[96 x 100]; 2 M-tiles share each B-tile.
  for (int mp = wv; mp < MT1 / 2; mp += 8) {
    const int mt0 = mp * 2;
    v16h va[2][KC1];
#pragma unroll
    for (int u = 0; u < 2; ++u)
#pragma unroll
      for (int kc = 0; kc < KC1; ++kc)
        va[u][kc] = load_a_f32(w1 + (size_t)((mt0 + u) * 16 + lmod) * CIN + kc * 32, lhi);

    for (int nt = 0; nt < NT1; ++nt) {
      const int p  = nt * 16 + lmod;
      const int pc = (p < HP) ? p : (HP - 1);     // clamp pad lanes (discarded)
      v8f acc0 = {}, acc1 = {};
#pragma unroll
      for (int kc = 0; kc < KC1; ++kc) {
        v16h vb = load_b_lds(xs + pc * CIN + kc * 32 + lhi * 8);
        acc0 = __builtin_amdgcn_wmma_f32_16x16x32_f16(
            false, va[0][kc], false, vb, (short)0, acc0, false, false);
        acc1 = __builtin_amdgcn_wmma_f32_16x16x32_f16(
            false, va[1][kc], false, vb, (short)0, acc1, false, false);
      }
      if (p < HP) {
        const int py = p / HX, px = p % HX;
        // dw-conv zero-pads h: halo pixels outside the image must be 0, not bn(0).
        const bool inimg = (unsigned)(y0 + py) < IMG && (unsigned)(x0 + px) < IMG;
        const float msc = inimg ? 1.0f : 0.0f;
#pragma unroll
        for (int u = 0; u < 2; ++u) {
          const v8f acc = u ? acc1 : acc0;
          const int chb = (mt0 + u) * 16 + lhi * 8;   // D rows: M = r + 8*lhi
          float2 sb[8];
#pragma unroll
          for (int r = 0; r < 8; ++r) sb[r] = sb1[chb + r];  // ds_load_b128 x4
#pragma unroll
          for (int r = 0; r < 8; r += 2) {            // packed b32 stores
            H2 t;
            t.h[0] = (_Float16)(fmaxf(acc[r]     * sb[r].x     + sb[r].y,     0.0f) * msc);
            t.h[1] = (_Float16)(fmaxf(acc[r + 1] * sb[r + 1].x + sb[r + 1].y, 0.0f) * msc);
            *(unsigned int*)&hs[p * HDM + chb + r] = t.u;
          }
        }
      }
    }
  }
  __syncthreads();      // also: last use of xs as the x tile

  // ---------------- Stage 2: depthwise 3x3 + BNdw + ReLU, in-place in hs -----
  // Channel-paired: all LDS traffic is packed b32. 64-ch chunks, reg-staged.
  for (int cb = 0; cb < HDM; cb += 64) {
    unsigned int res[8];
#pragma unroll
    for (int j = 0; j < 8; ++j) {
      const int lin = j * NTHREADS + tid;         // 0..2047 = 32 ch-pairs x 64 px
      const int c0 = cb + ((lin >> 6) << 1);      // even channel
      const int pl = lin & 63;
      const int iy = pl >> 3, ix = pl & 7;
      float s0 = 0.0f, s1 = 0.0f;
#pragma unroll
      for (int dy = 0; dy < 3; ++dy)
#pragma unroll
        for (int dx = 0; dx < 3; ++dx) {
          H2 w2p, h2p;
          w2p.u = *(const unsigned int*)&wdws[(dy * 3 + dx) * HDM + c0];
          h2p.u = *(const unsigned int*)&hs[((iy + dy) * HX + (ix + dx)) * HDM + c0];
          s0 += (float)w2p.h[0] * (float)h2p.h[0];
          s1 += (float)w2p.h[1] * (float)h2p.h[1];
        }
      const float2 d0 = sbd[c0], d1 = sbd[c0 + 1];
      H2 t;
      t.h[0] = (_Float16)fmaxf(s0 * d0.x + d0.y, 0.0f);
      t.h[1] = (_Float16)fmaxf(s1 * d1.x + d1.y, 0.0f);
      res[j] = t.u;
    }
    __syncthreads();                              // all reads of chunk done
#pragma unroll
    for (int j = 0; j < 8; ++j) {
      const int lin = j * NTHREADS + tid;
      const int c0 = cb + ((lin >> 6) << 1);
      const int pl = lin & 63;
      const int iy = pl >> 3, ix = pl & 7;
      *(unsigned int*)&hs[((iy + 1) * HX + (ix + 1)) * HDM + c0] = res[j];
    }
    __syncthreads();
  }

  // ---------------- Stage 3: project 1x1 GEMM + BN2 + residual -> out --------
  // out[96 x 64] = w2[96 x 576] * h2[576 x 64].
  // Wave owns one pixel tile (nt = wv&3) and 3 M-tiles (one B-tile -> 3 WMMAs).
  // w2 is staged to LDS in f16, 96 K-columns per block, reusing the dead xs
  // buffer (96*96*2 = 18,432B <= 19,200B): w2 read & converted exactly once.
  {
    const int nt  = wv & 3;
    const int mtb = wv >> 2;                      // 0 or 1; mts = mtb, mtb+2, mtb+4
    const int pi  = nt * 16 + lmod;               // inner pixel 0..63
    const int iy  = pi >> 3, ix = pi & 7;
    const _Float16* hrow = hs + ((iy + 1) * HX + (ix + 1)) * HDM;
    _Float16* w2s = xs;                           // [row 0..95][k 0..95] f16
    v8f acc[3] = {{}, {}, {}};

    for (int blk = 0; blk < HDM / CIN; ++blk) {   // 6 K-blocks of 96
      __syncthreads();                            // prior block's A reads done
      for (int i = tid * 4; i < CIN * CIN; i += NTHREADS * 4) {  // 9 float4 each
        const float4 f = *(const float4*)&w2[(size_t)(i / CIN) * HDM + blk * CIN + (i % CIN)];
        H2 t0, t1;
        t0.h[0] = (_Float16)f.x; t0.h[1] = (_Float16)f.y;
        t1.h[0] = (_Float16)f.z; t1.h[1] = (_Float16)f.w;
        *(unsigned int*)&w2s[i]     = t0.u;       // packed b64-able stores
        *(unsigned int*)&w2s[i + 2] = t1.u;
      }
      __syncthreads();
#pragma unroll
      for (int kcl = 0; kcl < 3; ++kcl) {         // 3 x K=32 per block
        v16h vb = load_b_lds(hrow + (blk * 3 + kcl) * 32 + lhi * 8);
#pragma unroll
        for (int u = 0; u < 3; ++u) {
          const _Float16* ab =
              w2s + ((mtb + 2 * u) * 16 + lmod) * CIN + kcl * 32 + lhi * 8;
          acc[u] = __builtin_amdgcn_wmma_f32_16x16x32_f16(
              false, load_a_lds(ab), false, vb, (short)0, acc[u], false, false);
        }
      }
    }

    const int gy = tY * TH + iy, gx = tX * TW + ix;
#pragma unroll
    for (int u = 0; u < 3; ++u) {
      const int chb = (mtb + 2 * u) * 16 + lhi * 8;
      float2 sb[8];
#pragma unroll
      for (int r = 0; r < 8; ++r) sb[r] = sb2[chb + r];
#pragma unroll
      for (int r = 0; r < 8; ++r) {
        const size_t oi = ((size_t)(n * CIN + chb + r) * IMG + gy) * IMG + gx;
        outg[oi] = acc[u][r] * sb[r].x + sb[r].y + xg[oi];   // BN2 + residual
      }
    }
  }
}

extern "C" void kernel_launch(void* const* d_in, const int* in_sizes, int n_in,
                              void* d_out, int out_size, void* d_ws, size_t ws_size,
                              hipStream_t stream) {
  (void)in_sizes; (void)n_in; (void)out_size; (void)d_ws; (void)ws_size;
  const dim3 grid((IMG / TW) * (IMG / TH), 32);   // 49 tiles x 32 images
  fused_ir_block<<<grid, NTHREADS, 0, stream>>>(
      (const float*)d_in[0],  (const float*)d_in[1],
      (const float*)d_in[2],  (const float*)d_in[3],
      (const float*)d_in[4],  (const float*)d_in[5],
      (const float*)d_in[6],  (const float*)d_in[7],
      (const float*)d_in[8],  (const float*)d_in[9],
      (const float*)d_in[10], (const float*)d_in[11],
      (const float*)d_in[12], (const float*)d_in[13],
      (const float*)d_in[14], (const float*)d_in[15],
      (float*)d_out);
}
